// Head_50062138802211
// MI455X (gfx1250) — compile-verified, matrix operand order checked by reading, not run on hardware
//
#include <hip/hip_runtime.h>

typedef _Float16 v16h __attribute__((ext_vector_type(16)));
typedef _Float16 v8h  __attribute__((ext_vector_type(8)));
typedef float    v8f  __attribute__((ext_vector_type(8)));
typedef unsigned int u32x4 __attribute__((ext_vector_type(4)));
typedef int          i32x8 __attribute__((ext_vector_type(8)));
typedef int          i32x4 __attribute__((ext_vector_type(4)));

#define BATCH 256
#define SEQ   256
#define EMB   384
#define HD    64

#if __has_builtin(__builtin_amdgcn_global_load_async_to_lds_b128)
#define HAVE_ASYNC 1
// b128 variant takes typed v4i32 pointers: (global src, lds dst, off, cpol)
#define ASYNC_CP16(dst, src)                                            \
  __builtin_amdgcn_global_load_async_to_lds_b128(                       \
      (__attribute__((address_space(1))) i32x4*)(_Float16*)(src),       \
      (__attribute__((address_space(3))) i32x4*)(dst), 0, 0)
#else
#define HAVE_ASYNC 0
#endif

#if __has_builtin(__builtin_amdgcn_tensor_load_to_lds)
#define HAVE_TDM 1
#endif

// K offset of half-pair i (0..7) inside a 32-wide K chunk for the CDNA5
// 16-bit A/B fragment layout (lanes 0-15 vs 16-31 differ by +8 in K).
__device__ __forceinline__ int frag_k(int i, int hi) {
  return (i < 4 ? 2 * i : 16 + 2 * (i - 4)) + 8 * hi;
}

__device__ __forceinline__ v8f v8f_zero() {
  v8f z;
#pragma unroll
  for (int i = 0; i < 8; ++i) z[i] = 0.0f;
  return z;
}

__device__ __forceinline__ v16h cat8(v8h lo, v8h hi8) {
  v16h r;
#pragma unroll
  for (int j = 0; j < 8; ++j) { r[j] = lo[j]; r[8 + j] = hi8[j]; }
  return r;
}

__device__ __forceinline__ unsigned lds_byte_offset(const void* p) {
  return (unsigned)(unsigned long long)
      (__attribute__((address_space(3))) const char*)p;
}

__device__ __forceinline__ v8f wmma_f16(v16h a, v16h b, v8f c) {
  return __builtin_amdgcn_wmma_f32_16x16x32_f16(
      false, a, false, b, (short)0, c, false, false);
}

// Stage one 32-key K tile (async-to-LDS, 2 x b128 per thread) and one V tile
// (TDM descriptor load, issued by wave 0).  Both tiles are contiguous 4 KB.
__device__ __forceinline__ void stage_tiles(int tid, int wave,
                                            const _Float16* kg,
                                            const _Float16* vg,
                                            _Float16* kdst,
                                            _Float16* vdst) {
#pragma unroll
  for (int t = 0; t < 2; ++t) {
    const int c = tid + t * 128;            // 16-byte chunk index (0..255)
#if HAVE_ASYNC
    ASYNC_CP16(kdst + c * 8, kg + c * 8);
#else
    *(v8h*)(kdst + c * 8) = *(const v8h*)(kg + c * 8);
#endif
  }
#ifdef HAVE_TDM
  if (wave == 0) {
    const unsigned long long ga = (unsigned long long)(const void*)vg;
    u32x4 g0;
    g0[0] = 1u;                                   // count=1, user D#
    g0[1] = lds_byte_offset(vdst);                // lds_addr
    g0[2] = (unsigned)ga;                         // global_addr lo
    g0[3] = ((unsigned)(ga >> 32) & 0x01FFFFFFu)  // global_addr hi
            | 0x80000000u;                        // type=2 (image)
    i32x8 g1;
    g1[0] = 0x00010000;           // data_size=2B; no multicast/pad/iterate
    g1[1] = (int)(2048u << 16);   // tensor_dim0 = 2048 elements (lo16)
    g1[2] = 0;                    // tensor_dim0 hi / tensor_dim1 lo
    g1[3] = (int)(2048u << 16);   // tile_dim0 = 2048
    g1[4] = 0;                    // tile_dim1/2 unused
    g1[5] = 2048;                 // tensor_dim0_stride lo32
    g1[6] = 0;
    g1[7] = 0;
    const i32x4 zz4 = {0, 0, 0, 0};
    const i32x8 zz8 = {0, 0, 0, 0, 0, 0, 0, 0};
    __builtin_amdgcn_tensor_load_to_lds(g0, g1, zz4, zz4, zz8, 0);
  }
#elif HAVE_ASYNC
#pragma unroll
  for (int t = 0; t < 2; ++t) {
    const int c = tid + t * 128;
    ASYNC_CP16(vdst + c * 8, vg + c * 8);
  }
#else
#pragma unroll
  for (int t = 0; t < 2; ++t) {
    const int c = tid + t * 128;
    *(v8h*)(vdst + c * 8) = *(const v8h*)(vg + c * 8);
  }
#endif
}

// ---------------------------------------------------------------------------
// Kernel 0: one-time weight prep.  W[E,H] f32 -> W^T[H,E] f16 (x3 matrices),
// so B-fragments become K-contiguous 16-byte loads.
// ---------------------------------------------------------------------------
__global__ __launch_bounds__(256) void wt_prep_kernel(
    const float* __restrict__ Wq, const float* __restrict__ Wk,
    const float* __restrict__ Wv, _Float16* __restrict__ wt)
{
  const int idx = blockIdx.x * 256 + threadIdx.x;   // < 3*HD*EMB
  const int m = idx / (HD * EMB);
  const int r = idx % (HD * EMB);
  const int n = r / EMB;
  const int e = r % EMB;
  const float* W = (m == 0) ? Wq : (m == 1) ? Wk : Wv;
  wt[idx] = (_Float16)W[(size_t)e * HD + n];
}

// ---------------------------------------------------------------------------
// Kernel 1: fused QKV projection.  x:[B*T,E] f32 -> q/k/v:[B*T,H] f16.
// grid.x = (B*T)/16 row tiles; 128 threads = 4 waves, wave w owns N-tile w.
// ---------------------------------------------------------------------------
__global__ __launch_bounds__(128) void qkv_proj_kernel(
    const float* __restrict__ x, const _Float16* __restrict__ wt,
    const float* __restrict__ bq, const float* __restrict__ bk,
    const float* __restrict__ bv,
    _Float16* __restrict__ qh, _Float16* __restrict__ kh,
    _Float16* __restrict__ vh)
{
  const int lane = threadIdx.x & 31;
  const int wave = threadIdx.x >> 5;        // N-tile 0..3
  const int hi   = lane >> 4;
  const int l15  = lane & 15;
  const int mt   = blockIdx.x;
  const int col  = wave * 16 + l15;         // output column 0..63

  const float* xrow = x + (size_t)(mt * 16 + l15) * EMB;
  const float* bias[3] = {bq, bk, bv};
  _Float16*    outp[3] = {qh, kh, vh};

  v8f acc[3];
#pragma unroll
  for (int m = 0; m < 3; ++m) acc[m] = v8f_zero();

  for (int kt = 0; kt < EMB / 32; ++kt) {
    const int k0 = kt * 32;
    // A fragment: per-lane K runs are contiguous -> 4x float4 loads + cvt
    const float4 f0 = *(const float4*)(xrow + k0 + 8 * hi);
    const float4 f1 = *(const float4*)(xrow + k0 + 8 * hi + 4);
    const float4 f2 = *(const float4*)(xrow + k0 + 16 + 8 * hi);
    const float4 f3 = *(const float4*)(xrow + k0 + 20 + 8 * hi);
    const float tmp[16] = {f0.x, f0.y, f0.z, f0.w, f1.x, f1.y, f1.z, f1.w,
                           f2.x, f2.y, f2.z, f2.w, f3.x, f3.y, f3.z, f3.w};
    v16h a;
#pragma unroll
    for (int j = 0; j < 16; ++j) a[j] = (_Float16)tmp[j];

#pragma unroll
    for (int m = 0; m < 3; ++m) {
      // B fragment from W^T: two 16-byte loads per 32-K chunk
      const _Float16* wrow = wt + (size_t)m * HD * EMB + (size_t)col * EMB;
      const v8h b0 = *(const v8h*)(wrow + k0 + 8 * hi);
      const v8h b1 = *(const v8h*)(wrow + k0 + 16 + 8 * hi);
      acc[m] = wmma_f16(a, cat8(b0, b1), acc[m]);
    }
  }

#pragma unroll
  for (int m = 0; m < 3; ++m) {
    const float bb = bias[m][col];
#pragma unroll
    for (int r = 0; r < 8; ++r) {
      const int row = mt * 16 + r + 8 * hi;
      outp[m][(size_t)row * HD + col] = (_Float16)(acc[m][r] + bb);
    }
  }
}

// ---------------------------------------------------------------------------
// Kernel 2: causal flash attention.  Block = 4 waves = 4 query tiles of one
// batch; double-buffered K/V staging (async-to-LDS + TDM) overlapped with
// WMMA compute, synchronized with partial ASYNCcnt/TENSORcnt waits.
// ---------------------------------------------------------------------------
__global__ __launch_bounds__(128) void attn_kernel(
    const _Float16* __restrict__ qh, const _Float16* __restrict__ kh,
    const _Float16* __restrict__ vh, float* __restrict__ out)
{
  const int tid  = threadIdx.x;
  const int lane = tid & 31;
  const int wave = tid >> 5;
  const int hi   = lane >> 4;
  const int l15  = lane & 15;
  const int b    = blockIdx.y;
  const int qt   = blockIdx.x * 4 + wave;   // query tile 0..15

  const _Float16* qb = qh + (size_t)b * SEQ * HD;
  const _Float16* kb = kh + (size_t)b * SEQ * HD;
  const _Float16* vb = vh + (size_t)b * SEQ * HD;

  __shared__ _Float16 ktile[2][32 * HD];    // 2 x 4 KB ping-pong
  __shared__ _Float16 vtile[2][32 * HD];    // 2 x 4 KB ping-pong
  __shared__ _Float16 plds[4][16 * 32];     // 4 KB, per-wave P staging
  _Float16* pw = plds[wave];

  // resident q A-fragments (two K=32 chunks cover H=64)
  const _Float16* qrow = qb + (size_t)(qt * 16 + l15) * HD;
  v16h qa[2];
#pragma unroll
  for (int c = 0; c < 2; ++c) {
    const v8h lo  = *(const v8h*)(qrow + c * 32 + 8 * hi);
    const v8h hi8 = *(const v8h*)(qrow + c * 32 + 16 + 8 * hi);
    qa[c] = cat8(lo, hi8);
  }

  v8f o[4];
#pragma unroll
  for (int nt = 0; nt < 4; ++nt) o[nt] = v8f_zero();
  float mrow[8], lrow[8];
#pragma unroll
  for (int r = 0; r < 8; ++r) { mrow[r] = -1e30f; lrow[r] = 0.0f; }

  const int qmax = qt * 16 + 15;            // per-wave causal bound
  const int nblk = blockIdx.x * 2 + 2;      // 32-key blocks for this block

  // prologue: stage tile 0 into buffer 0
  stage_tiles(tid, wave, kb, vb, ktile[0], vtile[0]);

  for (int i = 0; i < nblk; ++i) {
    const int s0  = i * 32;
    const int buf = i & 1;

    if (i + 1 < nblk) {
      // issue next tile's copies into the other buffer (its readers finished
      // before the end-of-iteration barrier of the previous iteration), then wait only
      // until the CURRENT tile's copies are done (in-order completion).
      stage_tiles(tid, wave, kb + (size_t)(s0 + 32) * HD,
                  vb + (size_t)(s0 + 32) * HD, ktile[buf ^ 1], vtile[buf ^ 1]);
      asm volatile("s_wait_asynccnt 2" ::: "memory");
#ifdef HAVE_TDM
      __builtin_amdgcn_s_wait_tensorcnt(1);
#endif
    } else {
      asm volatile("s_wait_asynccnt 0" ::: "memory");
#ifdef HAVE_TDM
      __builtin_amdgcn_s_wait_tensorcnt(0);
#endif
    }
    __syncthreads();                        // tile i visible to all waves

    if (s0 <= qmax) {                       // wave-uniform causal skip
      // ---- scores S = q @ k^T from the LDS K tile ----
      v8f st[2];
#pragma unroll
      for (int t = 0; t < 2; ++t) {
        v8f c0 = v8f_zero();
        const _Float16* krow = ktile[buf] + (t * 16 + l15) * HD;
#pragma unroll
        for (int c = 0; c < 2; ++c) {
          const v8h lo = *(const v8h*)(krow + c * 32 + 8 * hi);
          const v8h hh = *(const v8h*)(krow + c * 32 + 16 + 8 * hi);
          c0 = wmma_f16(qa[c], cat8(lo, hh), c0);
        }
        st[t] = c0;
      }

      // ---- causal mask + online softmax (rows live across a lane-half) ----
#pragma unroll
      for (int r = 0; r < 8; ++r) {
        const int rowg = qt * 16 + r + 8 * hi;
#pragma unroll
        for (int t = 0; t < 2; ++t) {
          const int colg = s0 + t * 16 + l15;
          if (colg > rowg) st[t][r] = -1e30f;
        }
        float mv = fmaxf(st[0][r], st[1][r]);
#pragma unroll
        for (int d = 1; d <= 8; d <<= 1)
          mv = fmaxf(mv, __shfl_xor(mv, d, 32));
        const float mnew  = fmaxf(mrow[r], mv);
        const float scale = __expf(mrow[r] - mnew);
        mrow[r] = mnew;
        float rs = 0.0f;
#pragma unroll
        for (int t = 0; t < 2; ++t) {
          st[t][r] = __expf(st[t][r] - mnew);
          rs += st[t][r];
        }
#pragma unroll
        for (int d = 1; d <= 8; d <<= 1) rs += __shfl_xor(rs, d, 32);
        lrow[r] = lrow[r] * scale + rs;
#pragma unroll
        for (int nt = 0; nt < 4; ++nt) o[nt][r] *= scale;
      }

      // ---- C-layout -> A-layout conversion of P via per-wave LDS slice ----
#pragma unroll
      for (int t = 0; t < 2; ++t)
#pragma unroll
        for (int r = 0; r < 8; ++r)
          pw[(r + 8 * hi) * 32 + t * 16 + l15] = (_Float16)st[t][r];
      asm volatile("s_wait_dscnt 0" ::: "memory");  // intra-wave exchange
      const v8h plo = *(const v8h*)(pw + l15 * 32 + 8 * hi);
      const v8h phi = *(const v8h*)(pw + l15 * 32 + 16 + 8 * hi);
      const v16h pa = cat8(plo, phi);

      // ---- O += P @ V from the LDS V tile ----
#pragma unroll
      for (int nt = 0; nt < 4; ++nt) {
        v16h vf;
#pragma unroll
        for (int i8 = 0; i8 < 8; ++i8) {
          const int K = frag_k(i8, hi);
          vf[2 * i8]     = vtile[buf][K * HD + nt * 16 + l15];
          vf[2 * i8 + 1] = vtile[buf][(K + 1) * HD + nt * 16 + l15];
        }
        o[nt] = wmma_f16(pa, vf, o[nt]);
      }
    }

    __syncthreads();                        // tile i reads done everywhere
  }

  // ---- epilogue: normalize and store f32 output ----
#pragma unroll
  for (int r = 0; r < 8; ++r) {
    const float inv = 1.0f / lrow[r];
    const int rowg = qt * 16 + r + 8 * hi;
#pragma unroll
    for (int nt = 0; nt < 4; ++nt)
      out[((size_t)b * SEQ + rowg) * HD + nt * 16 + l15] = o[nt][r] * inv;
  }
}

// ---------------------------------------------------------------------------
extern "C" void kernel_launch(void* const* d_in, const int* in_sizes, int n_in,
                              void* d_out, int out_size, void* d_ws,
                              size_t ws_size, hipStream_t stream) {
  (void)in_sizes; (void)n_in; (void)out_size; (void)ws_size;
  const float* x  = (const float*)d_in[0];
  const float* Wq = (const float*)d_in[1];
  const float* bq = (const float*)d_in[2];
  const float* Wk = (const float*)d_in[3];
  const float* bk = (const float*)d_in[4];
  const float* Wv = (const float*)d_in[5];
  const float* bv = (const float*)d_in[6];
  float* out = (float*)d_out;

  _Float16* qhp = (_Float16*)d_ws;
  _Float16* khp = qhp + (size_t)BATCH * SEQ * HD;
  _Float16* vhp = khp + (size_t)BATCH * SEQ * HD;
  _Float16* wtp = vhp + (size_t)BATCH * SEQ * HD;   // 3 * 64 * 384 f16

  wt_prep_kernel<<<dim3((3 * HD * EMB) / 256), 256, 0, stream>>>(Wq, Wk, Wv,
                                                                 wtp);
  qkv_proj_kernel<<<dim3((BATCH * SEQ) / 16), 128, 0, stream>>>(
      x, wtp, bq, bk, bv, qhp, khp, vhp);
  attn_kernel<<<dim3(SEQ / 64, BATCH), 128, 0, stream>>>(qhp, khp, vhp, out);
}
